// GNNWithOptionalMoE_10943576670417
// MI455X (gfx1250) — matrix operand order repaired and checked by reading, not compiled, not run on hardware
//
#include <hip/hip_runtime.h>
#include <hip/hip_bf16.h>
#include <math.h>

// ---------------------------------------------------------------------------
// MoE FFN (top-2 of 4 experts), d_model=1024, tokens=16384.
// bf16 WMMA for both GEMMs; LDS-staged operand slabs to remove cross-wave
// duplicate global loads; f32 epilogues (erf-GELU, residual+LayerNorm).
// ---------------------------------------------------------------------------

#define N_TOKENS 16384
#define D_MODEL  1024
#define D_FF     2048     // 2*d
#define N_EXP    4
#define KSTEP    64
#define LPAD     72       // padded LDS row stride in elements (64 + 8)

typedef __attribute__((ext_vector_type(16))) __bf16 v16bf;
typedef __attribute__((ext_vector_type(8)))  __bf16 v8bf;
typedef __attribute__((ext_vector_type(8)))  float  v8f;

// ---------------- fragment loaders (CDNA5 16-bit WMMA layouts) -------------
// Works for both global and LDS pointers (addrspace inferred by compiler).
// A matrix 16x32 (MxK): lane<16 holds row M=lane, K {k0..k0+7, k0+16..k0+23};
// lane>=16 holds row M=lane-16, K {k0+8..k0+15, k0+24..k0+31}.
__device__ __forceinline__ v16bf frag_A(const __bf16* base, int stride,
                                        int row0, int k0) {
  const int lane = threadIdx.x & 31;
  const int lr   = lane & 15;
  const int half = lane >> 4;
  const __bf16* p = base + (size_t)(row0 + lr) * stride + k0 + half * 8;
  v8bf lo = *reinterpret_cast<const v8bf*>(p);
  v8bf hi = *reinterpret_cast<const v8bf*>(p + 16);
  v16bf r;
#pragma unroll
  for (int i = 0; i < 8; ++i) { r[i] = lo[i]; r[i + 8] = hi[i]; }
  return r;
}

// B matrix 32x16 (KxN): lane<16 holds col N=lane, K {k0..k0+7, k0+16..k0+23};
// lane>=16 holds col N=lane-16, K {k0+8..k0+15, k0+24..k0+31}. Source row-major W[j][k].
__device__ __forceinline__ v16bf frag_B(const __bf16* base, int stride,
                                        int col0, int k0) {
  const int lane = threadIdx.x & 31;
  const int lr   = lane & 15;
  const int half = lane >> 4;
  const __bf16* p = base + (size_t)(col0 + lr) * stride + k0 + half * 8;
  v8bf lo = *reinterpret_cast<const v8bf*>(p);
  v8bf hi = *reinterpret_cast<const v8bf*>(p + 16);
  v16bf r;
#pragma unroll
  for (int i = 0; i < 8; ++i) { r[i] = lo[i]; r[i + 8] = hi[i]; }
  return r;
}

__device__ __forceinline__ v8f wmma_bf16(v16bf a, v16bf b, v8f c) {
  return __builtin_amdgcn_wmma_f32_16x16x32_bf16(
      false, a, false, b, (short)0, c, false, false);
}

__device__ __forceinline__ float gelu_exact(float v) {
  return 0.5f * v * (1.0f + erff(v * 0.70710678118654752440f));
}

// ---------------------------- convert f32 -> bf16 --------------------------
__global__ __launch_bounds__(256) void cvt_bf16_kernel(
    const float* __restrict__ src, __bf16* __restrict__ dst, size_t n) {
  for (size_t i = (size_t)blockIdx.x * blockDim.x + threadIdx.x; i < n;
       i += (size_t)gridDim.x * blockDim.x) {
    dst[i] = (__bf16)src[i];
  }
}

// ------------------------------- gating ------------------------------------
__global__ __launch_bounds__(128) void gate_kernel(
    const float* __restrict__ x, const float* __restrict__ gw,
    float* __restrict__ comb) {
  const int wid   = threadIdx.x >> 5;
  const int lane  = threadIdx.x & 31;
  const int token = blockIdx.x * 4 + wid;
  if (token >= N_TOKENS) return;

  float s0 = 0.f, s1 = 0.f, s2 = 0.f, s3 = 0.f;
  const float* xp = x + (size_t)token * D_MODEL;
  for (int k = lane; k < D_MODEL; k += 32) {
    const float xv = xp[k];
    s0 += xv * gw[0 * D_MODEL + k];
    s1 += xv * gw[1 * D_MODEL + k];
    s2 += xv * gw[2 * D_MODEL + k];
    s3 += xv * gw[3 * D_MODEL + k];
  }
#pragma unroll
  for (int off = 16; off >= 1; off >>= 1) {
    s0 += __shfl_xor(s0, off, 32);
    s1 += __shfl_xor(s1, off, 32);
    s2 += __shfl_xor(s2, off, 32);
    s3 += __shfl_xor(s3, off, 32);
  }
  if (lane == 0) {
    float s[4] = {s0, s1, s2, s3};
    int i0 = 0;
    for (int e = 1; e < 4; ++e) if (s[e] > s[i0]) i0 = e;
    int i1 = (i0 == 0) ? 1 : 0;
    for (int e = 0; e < 4; ++e)
      if (e != i0 && s[e] > s[i1]) i1 = e;
    const float p1 = __expf(s[i1] - s[i0]);
    const float inv = 1.0f / (1.0f + p1);
    float out[4] = {0.f, 0.f, 0.f, 0.f};
    out[i0] = inv;
    out[i1] = p1 * inv;
    float* cp = comb + (size_t)token * N_EXP;
    cp[0] = out[0]; cp[1] = out[1]; cp[2] = out[2]; cp[3] = out[3];
  }
}

// ------------------------- FFN1: H = gelu(x W1^T + b1) ---------------------
// Block: 256 threads = 8 waves. Block tile: 128 rows x 128 cols.
// Per K-step(64): cooperatively stage A(128x64) and B(128x64) slabs in LDS,
// then each wave (2 M-tiles x 4 N-tiles) runs 16 WMMAs from LDS fragments.
__global__ __launch_bounds__(256) void ffn1_kernel(
    const __bf16* __restrict__ xb, const __bf16* __restrict__ w1b,
    const float* __restrict__ b1, __bf16* __restrict__ H, int expert) {
  __shared__ __bf16 As[128 * LPAD];   // 18 KB
  __shared__ __bf16 Bs[128 * LPAD];   // 18 KB

  const int tid  = threadIdx.x;
  const int wave = tid >> 5;
  const int lane = tid & 31;
  const int wr   = wave & 3;           // 0..3 -> M sub-tile group
  const int wc   = wave >> 2;          // 0..1 -> N sub-tile group
  const int mb   = blockIdx.y * 128;   // block row base (tokens)
  const int nb   = blockIdx.x * 128;   // block col base (ff)

  const __bf16* w1e = w1b + (size_t)expert * D_FF * D_MODEL;
  const float*  b1e = b1 + (size_t)expert * D_FF;

  v8f acc[2][4] = {};
  for (int k0 = 0; k0 < D_MODEL; k0 += KSTEP) {
    // cooperative staging: 1024 chunks of 16B each for A and B
#pragma unroll
    for (int i = 0; i < 4; ++i) {
      const int c   = i * 256 + tid;        // 0..1023
      const int row = c >> 3;               // 0..127
      const int seg = c & 7;                // 8 elems per 16B seg
      *reinterpret_cast<v8bf*>(&As[row * LPAD + seg * 8]) =
          *reinterpret_cast<const v8bf*>(
              xb + (size_t)(mb + row) * D_MODEL + k0 + seg * 8);
      *reinterpret_cast<v8bf*>(&Bs[row * LPAD + seg * 8]) =
          *reinterpret_cast<const v8bf*>(
              w1e + (size_t)(nb + row) * D_MODEL + k0 + seg * 8);
    }
    __syncthreads();

#pragma unroll
    for (int kk = 0; kk < KSTEP; kk += 32) {
      v16bf a0 = frag_A(As, LPAD, wr * 32, kk);
      v16bf a1 = frag_A(As, LPAD, wr * 32 + 16, kk);
#pragma unroll
      for (int t = 0; t < 4; ++t) {
        v16bf b = frag_B(Bs, LPAD, wc * 64 + t * 16, kk);
        acc[0][t] = wmma_bf16(a0, b, acc[0][t]);
        acc[1][t] = wmma_bf16(a1, b, acc[1][t]);
      }
    }
    __syncthreads();
  }

  // Epilogue: lane<16 -> rows +0..7, lane>=16 -> rows +8..15; col = n + lr.
  const int lr   = lane & 15;
  const int half = lane >> 4;
#pragma unroll
  for (int rt = 0; rt < 2; ++rt) {
#pragma unroll
    for (int t = 0; t < 4; ++t) {
      const int col = nb + wc * 64 + t * 16 + lr;
      const float bias = b1e[col];
#pragma unroll
      for (int r = 0; r < 8; ++r) {
        const int row = mb + wr * 32 + rt * 16 + half * 8 + r;
        const float v = gelu_exact(acc[rt][t][r] + bias);
        H[(size_t)row * D_FF + col] = (__bf16)v;
      }
    }
  }
}

// --------- FFN2: z = x + H W2^T + b2 ; LN(z); y (+)= comb * LN(z) ----------
// Block: 512 threads = 16 waves; owns 32 tokens x all 1024 columns.
// A (H slab, shared by ALL waves) staged in LDS; B direct from global
// (no cross-wave duplication). LayerNorm in two 16-row phases via zs.
__global__ __launch_bounds__(512) void ffn2_kernel(
    const __bf16* __restrict__ Hb, const __bf16* __restrict__ w2b,
    const float* __restrict__ b2, const float* __restrict__ x,
    const float* __restrict__ comb, const float* __restrict__ gamma,
    const float* __restrict__ beta, float* __restrict__ y,
    int expert, int first) {
  __shared__ float zs[16 * D_MODEL];   // 64 KB; front aliased as A staging
  __bf16* As = reinterpret_cast<__bf16*>(zs);  // needs 32*LPAD*2 = 4.6 KB

  const int tid  = threadIdx.x;
  const int wave = tid >> 5;           // 0..15
  const int lane = tid & 31;
  const int m0   = blockIdx.x * 32;

  const __bf16* w2e = w2b + (size_t)expert * D_MODEL * D_FF;
  const float*  b2e = b2 + (size_t)expert * D_MODEL;
  const float*  ge  = gamma + (size_t)expert * D_MODEL;
  const float*  be  = beta + (size_t)expert * D_MODEL;

  v8f acc[2][4] = {};
  for (int k0 = 0; k0 < D_FF; k0 += KSTEP) {
    // stage H slab: 32 rows x 64 elems = 256 chunks of 16B (threads 0..255)
    if (tid < 256) {
      const int row = tid >> 3;
      const int seg = tid & 7;
      *reinterpret_cast<v8bf*>(&As[row * LPAD + seg * 8]) =
          *reinterpret_cast<const v8bf*>(
              Hb + (size_t)(m0 + row) * D_FF + k0 + seg * 8);
    }
    if (k0 + KSTEP < D_FF) {  // prefetch next weight slab rows for this wave
      __builtin_prefetch(
          w2e + (size_t)(wave * 64 + (lane & 15)) * D_FF + k0 + KSTEP, 0, 3);
    }
    __syncthreads();

#pragma unroll
    for (int kk = 0; kk < KSTEP; kk += 32) {
      v16bf a0 = frag_A(As, LPAD, 0, kk);
      v16bf a1 = frag_A(As, LPAD, 16, kk);
#pragma unroll
      for (int t = 0; t < 4; ++t) {
        v16bf b = frag_B(w2e, D_FF, wave * 64 + t * 16, k0 + kk);
        acc[0][t] = wmma_bf16(a0, b, acc[0][t]);
        acc[1][t] = wmma_bf16(a1, b, acc[1][t]);
      }
    }
    __syncthreads();
  }

  const int lr   = lane & 15;
  const int half = lane >> 4;

#pragma unroll
  for (int ph = 0; ph < 2; ++ph) {
    // stage rows [m0 + ph*16, +16): acc + bias + residual -> zs
#pragma unroll
    for (int t = 0; t < 4; ++t) {
      const int col = wave * 64 + t * 16 + lr;
      const float bias = b2e[col];
#pragma unroll
      for (int r = 0; r < 8; ++r) {
        const int row = half * 8 + r;            // 0..15 within phase
        const int tok = m0 + ph * 16 + row;
        zs[row * D_MODEL + col] =
            acc[ph][t][r] + bias + x[(size_t)tok * D_MODEL + col];
      }
    }
    __syncthreads();

    // LayerNorm: wave w normalizes row w of this phase.
    const int row = wave;
    const int tok = m0 + ph * 16 + row;
    float s = 0.f, s2 = 0.f;
    for (int c = lane; c < D_MODEL; c += 32) {
      const float v = zs[row * D_MODEL + c];
      s += v; s2 += v * v;
    }
#pragma unroll
    for (int off = 16; off >= 1; off >>= 1) {
      s  += __shfl_xor(s, off, 32);
      s2 += __shfl_xor(s2, off, 32);
    }
    const float mean = s * (1.0f / D_MODEL);
    const float var  = s2 * (1.0f / D_MODEL) - mean * mean;
    const float rstd = rsqrtf(var + 1e-6f);
    const float wgt  = comb[(size_t)tok * N_EXP + expert];

    for (int c = lane; c < D_MODEL; c += 32) {
      const float zh  = (zs[row * D_MODEL + c] - mean) * rstd;
      const float val = wgt * (zh * ge[c] + be[c]);
      const size_t idx = (size_t)tok * D_MODEL + c;
      if (first) y[idx] = val;
      else       y[idx] += val;
    }
    __syncthreads();   // protect zs before next phase overwrites it
  }
}

// ------------------------------- launcher ----------------------------------
extern "C" void kernel_launch(void* const* d_in, const int* in_sizes, int n_in,
                              void* d_out, int out_size, void* d_ws, size_t ws_size,
                              hipStream_t stream) {
  (void)in_sizes; (void)n_in; (void)out_size; (void)ws_size;

  const float* x      = (const float*)d_in[0];
  const float* gate_w = (const float*)d_in[1];
  const float* w1     = (const float*)d_in[2];
  const float* b1     = (const float*)d_in[3];
  const float* w2     = (const float*)d_in[4];
  const float* b2     = (const float*)d_in[5];
  const float* gamma  = (const float*)d_in[6];
  const float* beta   = (const float*)d_in[7];
  float* y = (float*)d_out;

  // workspace carve-up
  char* ws = (char*)d_ws;
  __bf16* xb  = (__bf16*)ws;  ws += (size_t)N_TOKENS * D_MODEL * 2;          // 32 MB
  __bf16* w1b = (__bf16*)ws;  ws += (size_t)N_EXP * D_FF * D_MODEL * 2;     // 16 MB
  __bf16* w2b = (__bf16*)ws;  ws += (size_t)N_EXP * D_MODEL * D_FF * 2;     // 16 MB
  float*  comb = (float*)ws;  ws += (size_t)N_TOKENS * N_EXP * 4;           // 256 KB
  __bf16* H   = (__bf16*)ws;  ws += (size_t)N_TOKENS * D_FF * 2;            // 64 MB

  // 1) convert to bf16
  cvt_bf16_kernel<<<4096, 256, 0, stream>>>(x,  xb,  (size_t)N_TOKENS * D_MODEL);
  cvt_bf16_kernel<<<2048, 256, 0, stream>>>(w1, w1b, (size_t)N_EXP * D_FF * D_MODEL);
  cvt_bf16_kernel<<<2048, 256, 0, stream>>>(w2, w2b, (size_t)N_EXP * D_MODEL * D_FF);

  // 2) gating (top-2 softmax -> dense comb[N,4])
  gate_kernel<<<N_TOKENS / 4, 128, 0, stream>>>(x, gate_w, comb);

  // 3) experts, sequential (H buffer reused; stream order gives dependency)
  for (int e = 0; e < N_EXP; ++e) {
    dim3 g1(D_FF / 128, N_TOKENS / 128);
    ffn1_kernel<<<g1, 256, 0, stream>>>(xb, w1b, b1, H, e);
    ffn2_kernel<<<N_TOKENS / 32, 512, 0, stream>>>(H, w2b, b2, x, comb, gamma,
                                                   beta, y, e, e == 0 ? 1 : 0);
  }
}
